// DGCNN_18760417149347
// MI455X (gfx1250) — matrix operand order repaired and checked by reading, not compiled, not run on hardware
//
#include <hip/hip_runtime.h>

#define DEV __device__ __forceinline__

typedef __attribute__((ext_vector_type(16))) _Float16 v16h;
typedef __attribute__((ext_vector_type(8)))  _Float16 v8h;
typedef __attribute__((ext_vector_type(8)))  float    v8f;

typedef __attribute__((ext_vector_type(4))) unsigned int tdm_u4;
typedef __attribute__((ext_vector_type(8))) int          tdm_i8;
typedef __attribute__((ext_vector_type(4))) int          tdm_i4;

static constexpr int B_   = 8;
static constexpr int N_   = 4096;
static constexpr int KNN  = 20;
static constexpr int CINP = 32;          // 26 padded to 32
static constexpr int PTS  = 8;           // points per edge-conv workgroup
static constexpr int NCOL = PTS * KNN;   // 160 columns
static constexpr int LDC  = 136;         // LDS column stride in halves (bank-friendly)
static constexpr int CATC = 512;

#if defined(__has_builtin)
#if __has_builtin(__builtin_amdgcn_tensor_load_to_lds)
#define HAVE_TDM 1
#endif
#endif
#ifndef HAVE_TDM
#define HAVE_TDM 0
#endif

// ---- WMMA fragment helpers (CDNA5 16x16x32 f16 layouts, ISA 7.12.2) ----

// A (MxK=16x32): lane<16 rows m, K {0..7,16..23}; lane>=16 same rows, K {8..15,24..31}
DEV v16h load_a_frag(const _Float16* W, int ldw, int mbase, int kb, int lane) {
  int m  = mbase + (lane & 15);
  int hb = ((lane >> 4) & 1) * 8;
  const _Float16* p = W + m * ldw + kb + hb;
  v8h lo = *(const v8h*)(p);
  v8h hi = *(const v8h*)(p + 16);
  v16h a;
#pragma unroll
  for (int i = 0; i < 8; ++i) { a[i] = lo[i]; a[i + 8] = hi[i]; }
  return a;
}

// B (KxN=32x16): lane<16 col n, K {0..7,16..23}; lane>=16 col n, K {8..15,24..31}
DEV v16h load_b_frag(const _Float16* Bm, int ldb, int nbase, int kb, int lane) {
  int col = nbase + (lane & 15);
  int kk  = kb + ((lane >> 4) & 1) * 8;
  const _Float16* p = Bm + col * ldb + kk;
  v8h lo = *(const v8h*)(p);
  v8h hi = *(const v8h*)(p + 16);
  v16h b;
#pragma unroll
  for (int i = 0; i < 8; ++i) { b[i] = lo[i]; b[i + 8] = hi[i]; }
  return b;
}

DEV v8f wmma_f16(v16h a, v16h b, v8f c) {
  return __builtin_amdgcn_wmma_f32_16x16x32_f16(false, a, false, b, (short)0, c,
                                                false, false);
}

// ---------------- kernel 0: weight conversion f32 -> f16 ----------------
__global__ void cvt_weights_kernel(const float* __restrict__ w1, const float* __restrict__ w2,
                                   const float* __restrict__ w3, const float* __restrict__ w4,
                                   const float* __restrict__ w5,
                                   _Float16* __restrict__ w1p, _Float16* __restrict__ w2h,
                                   _Float16* __restrict__ w3h, _Float16* __restrict__ w4h,
                                   _Float16* __restrict__ w5h) {
  int i = blockIdx.x * 256 + threadIdx.x;
  if (i < 64 * 32) {
    int o = i >> 5, c = i & 31;
    w1p[i] = (c < 26) ? (_Float16)w1[o * 26 + c] : (_Float16)0.0f;
  }
  if (i < 64 * 64)   w2h[i] = (_Float16)w2[i];
  if (i < 128 * 64)  w3h[i] = (_Float16)w3[i];
  if (i < 256 * 128) w4h[i] = (_Float16)w4[i];
  if (i < 512 * 512) w5h[i] = (_Float16)w5[i];
}

// ---------------- kernel 1: kNN (one wave32 per row) ----------------
__global__ void knn_kernel(const float* __restrict__ x, int* __restrict__ knn_idx,
                           _Float16* __restrict__ knn_dist) {
  extern __shared__ float smem[];          // 3*N floats + 8*640 float + 8*640 int
  float* xs    = smem;
  float* mvals = smem + 3 * N_;
  int*   minds = (int*)(mvals + 8 * 640);

  int tid = threadIdx.x, lane = tid & 31, wave = tid >> 5;
  int batch = blockIdx.x >> 9;             // 512 blocks per batch
  int row0  = (blockIdx.x & 511) * 8;

  for (int i = tid; i < 3 * N_; i += 256) xs[i] = x[batch * 3 * N_ + i];
  __syncthreads();

  int n = row0 + wave;
  float xn0 = xs[n], xn1 = xs[N_ + n], xn2 = xs[2 * N_ + n];

  float bv[KNN]; int bi[KNN];
#pragma unroll
  for (int j = 0; j < KNN; ++j) { bv[j] = -3.0e38f; bi[j] = 0; }

  for (int m = lane; m < N_; m += 32) {
    float d0 = xs[m] - xn0, d1 = xs[N_ + m] - xn1, d2 = xs[2 * N_ + m] - xn2;
    float d = -(d0 * d0 + d1 * d1 + d2 * d2);    // == pair value in reference
    if (d > bv[KNN - 1]) {
      bv[KNN - 1] = d; bi[KNN - 1] = m;
#pragma unroll
      for (int j = KNN - 1; j > 0; --j) {
        if (bv[j] > bv[j - 1]) {
          float tv = bv[j]; bv[j] = bv[j - 1]; bv[j - 1] = tv;
          int   ti = bi[j]; bi[j] = bi[j - 1]; bi[j - 1] = ti;
        }
      }
    }
  }

  float* mv = mvals + wave * 640;
  int*   mi = minds + wave * 640;
#pragma unroll
  for (int j = 0; j < KNN; ++j) { mv[lane * KNN + j] = bv[j]; mi[lane * KNN + j] = bi[j]; }

  long ob = ((long)batch * N_ + n) * KNN;
  for (int k = 0; k < KNN; ++k) {
    float best = -3.0e38f; int bpos = lane * KNN;
#pragma unroll
    for (int j = 0; j < KNN; ++j) {
      float v = mv[lane * KNN + j];
      if (v > best) { best = v; bpos = lane * KNN + j; }
    }
#pragma unroll
    for (int off = 16; off > 0; off >>= 1) {
      float ov = __shfl_xor(best, off, 32);
      int   op = __shfl_xor(bpos, off, 32);
      if (ov > best || (ov == best && op < bpos)) { best = ov; bpos = op; }
    }
    if (lane == 0) {
      knn_idx[ob + k]  = mi[bpos];
      knn_dist[ob + k] = (_Float16)sqrtf(fmaxf(-best, 0.0f) + 1e-12f);
      mv[bpos] = -3.0e38f;
    }
  }
}

// ---------------- kernel 2: fused edge-conv stack (WMMA, LDS-resident) ----------------
__global__ void edge_kernel(const float* __restrict__ x,
                            const int* __restrict__ knn_idx,
                            const _Float16* __restrict__ knn_dist,
                            const _Float16* __restrict__ w1p, const _Float16* __restrict__ w2h,
                            const _Float16* __restrict__ w3h, const _Float16* __restrict__ w4h,
                            _Float16* __restrict__ cat) {
  extern __shared__ _Float16 hbuf[];       // 2 * NCOL * LDC halves
  _Float16* bufX = hbuf;
  _Float16* bufY = hbuf + NCOL * LDC;

  int tid = threadIdx.x, lane = tid & 31, wave = tid >> 5;
  int b  = blockIdx.x >> 9;
  int p0 = (blockIdx.x & 511) * PTS;

  // ---- build feature columns into bufX channels [0,32), packed b128 stores
  for (int col = tid; col < NCOL; col += 256) {
    int p = col / KNN, kk = col % KNN;
    int n = p0 + p;
    long nb = ((long)b * N_ + n) * KNN;
    int gi = knn_idx[nb + kk];
    _Float16 dv[KNN];
#pragma unroll
    for (int j = 0; j < KNN; ++j) dv[j] = knn_dist[nb + j];
    v8h f0, f1, f2, f3;
#pragma unroll
    for (int d = 0; d < 3; ++d) {
      f0[d]     = (_Float16)x[(b * 3 + d) * N_ + n];
      f0[3 + d] = (_Float16)x[(b * 3 + d) * N_ + gi];
    }
    f0[6] = dv[0]; f0[7] = dv[1];
#pragma unroll
    for (int r = 0; r < 8; ++r) f1[r] = dv[2 + r];
#pragma unroll
    for (int r = 0; r < 8; ++r) f2[r] = dv[10 + r];
    f3[0] = dv[18]; f3[1] = dv[19];
#pragma unroll
    for (int r = 2; r < 8; ++r) f3[r] = (_Float16)0.0f;
    _Float16* dst = bufX + col * LDC;
    *(v8h*)(dst)      = f0;
    *(v8h*)(dst + 8)  = f1;
    *(v8h*)(dst + 16) = f2;
    *(v8h*)(dst + 24) = f3;
  }
  __syncthreads();

  auto conv = [&](const _Float16* W, int ldw, int mtiles, int ksteps,
                  int mrow_off, const _Float16* src, _Float16* dst) {
    const int ntiles = NCOL / 16;          // 10
    for (int t = wave; t < mtiles * ntiles; t += 8) {
      int mo = t / ntiles, no = t % ntiles;
      v8f acc = {};
      for (int ks = 0; ks < ksteps; ++ks) {
        v16h a  = load_a_frag(W, ldw, (mrow_off + mo) * 16, ks * 32, lane);
        v16h bb = load_b_frag(src, LDC, no * 16, ks * 32, lane);
        acc = wmma_f16(a, bb, acc);
      }
      int col = no * 16 + (lane & 15);
      int cb  = mo * 16 + ((lane >> 4) & 1) * 8;
      v8h o;
#pragma unroll
      for (int r = 0; r < 8; ++r) o[r] = (_Float16)fmaxf(acc[r], 0.0f);
      *(v8h*)(dst + col * LDC + cb) = o;   // relu'd f16, ds_store_b128
    }
  };

  auto pool = [&](const _Float16* buf, int C, int coff) {
    int c8n = C >> 3;                       // 8-channel groups
    for (int i = tid; i < PTS * c8n; i += 256) {
      int p = i / c8n, c8 = (i % c8n) * 8;
      v8h mx = *(const v8h*)(buf + (p * KNN) * LDC + c8);
      for (int kk = 1; kk < KNN; ++kk) {
        v8h v = *(const v8h*)(buf + (p * KNN + kk) * LDC + c8);
#pragma unroll
        for (int r = 0; r < 8; ++r) mx[r] = (v[r] > mx[r]) ? v[r] : mx[r];
      }
      *(v8h*)(cat + ((long)(b * N_ + p0 + p)) * CATC + coff + c8) = mx;
    }
  };

  conv(w1p, 32, 4, 1, 0, bufX, bufY);  __syncthreads();
  pool(bufY, 64, 0);
  conv(w2h, 64, 4, 2, 0, bufY, bufX);  __syncthreads();
  pool(bufX, 64, 64);
  conv(w3h, 64, 8, 2, 0, bufX, bufY);  __syncthreads();
  pool(bufY, 128, 128);
  conv(w4h, 128, 8, 4, 0, bufY, bufX); __syncthreads();   // w4 rows 0..127
  pool(bufX, 128, 256);
  __syncthreads();                                        // protect bufX readers
  conv(w4h, 128, 8, 4, 8, bufY, bufX); __syncthreads();   // w4 rows 128..255
  pool(bufX, 128, 384);
}

// ---------------- kernel 3: final 512x512 GEMM + relu, TDM-staged B ----------------
// LDS: 2 buffers, each 128 cols x (32 data + 8 pad) halves = 80B column stride
static constexpr int LDB  = 40;            // halves (32 data + 4-dword TDM pad)
static constexpr int BUFH = 128 * LDB;     // halves per buffer
static constexpr unsigned BUFB = BUFH * 2; // bytes per buffer

#if HAVE_TDM
DEV void tdm_load_cat_tile(const _Float16* cat, long col0, int kb, unsigned lds_off) {
  unsigned long long ga = (unsigned long long)(const void*)(cat + col0 * CATC + kb);
  tdm_u4 g0;
  g0[0] = 1u;                                            // count=1 valid user D#
  g0[1] = lds_off;                                       // lds_addr (bytes)
  g0[2] = (unsigned)(ga & 0xffffffffu);                  // global_addr[31:0]
  g0[3] = (unsigned)((ga >> 32) & 0x01ffffffu) | (2u << 30);  // addr[56:32], type=2
  tdm_i8 g1;
  g1[0] = (int)((1u << 16) | (1u << 20) | (3u << 22) | (3u << 25));
  //            data_size=2B  pad_en      every 16 dw    pad 4 dw
  g1[1] = (int)(((unsigned)CATC & 0xffffu) << 16);       // tensor_dim0 = 512 (lo16)
  g1[2] = (int)(((unsigned)(B_ * N_) & 0xffffu) << 16);  // dim0 hi=0 | tensor_dim1 lo16
  g1[3] = (int)(32u << 16);                              // dim1 hi=0 | tile_dim0 = 32
  g1[4] = 128;                                           // tile_dim1 = 128 cols
  g1[5] = CATC;                                          // tensor_dim0_stride = 512
  g1[6] = 0;
  g1[7] = 0;
  tdm_i4 z4 = (tdm_i4)0;
#if __clang_major__ >= 23
  tdm_i8 z8 = (tdm_i8)0;
  __builtin_amdgcn_tensor_load_to_lds(g0, g1, z4, z4, z8, 0);
#else
  __builtin_amdgcn_tensor_load_to_lds(g0, g1, z4, z4, 0);
#endif
}
#endif

__global__ void gemm5_kernel(const _Float16* __restrict__ w5h,
                             const _Float16* __restrict__ cat,
                             float* __restrict__ out) {
  extern __shared__ _Float16 blds[];       // 2 * BUFH halves, starts at LDS offset 0
  int tid = threadIdx.x, lane = tid & 31, wave = tid >> 5;
  int mblk = blockIdx.x & 7;               // 8 M blocks of 64 rows
  int nblk = blockIdx.x >> 3;              // 256 N blocks of 128 cols
  int mbase  = mblk * 64 + (wave & 1) * 32;
  long nbase0 = (long)nblk * 128;
  int nloc   = (wave >> 1) * 32;           // wave's column offset inside LDS tile

#if HAVE_TDM
  if (wave == 0) {
    tdm_load_cat_tile(cat, nbase0, 0, 0);
    __builtin_amdgcn_s_wait_tensorcnt(0);
  }
  __syncthreads();
#else
  for (int i = tid; i < 128 * 4; i += 256) {
    int col = i >> 2, part = i & 3;
    *(v8h*)(blds + col * LDB + part * 8) =
        *(const v8h*)(cat + (nbase0 + col) * CATC + part * 8);
  }
  __syncthreads();
#endif

  v8f a00 = {}, a01 = {}, a10 = {}, a11 = {};
  int cur = 0;
  for (int kb = 0; kb < 512; kb += 32, cur ^= 1) {
#if HAVE_TDM
    if (wave == 0 && kb + 32 < 512)
      tdm_load_cat_tile(cat, nbase0, kb + 32, (unsigned)(cur ^ 1) * BUFB);
#endif
    const _Float16* bcur = blds + cur * BUFH;
    v16h fa0 = load_a_frag(w5h, 512, mbase,      kb, lane);
    v16h fa1 = load_a_frag(w5h, 512, mbase + 16, kb, lane);
    v16h fb0 = load_b_frag(bcur, LDB, nloc,      0, lane);
    v16h fb1 = load_b_frag(bcur, LDB, nloc + 16, 0, lane);
    a00 = wmma_f16(fa0, fb0, a00);
    a01 = wmma_f16(fa0, fb1, a01);
    a10 = wmma_f16(fa1, fb0, a10);
    a11 = wmma_f16(fa1, fb1, a11);
#if HAVE_TDM
    if (wave == 0) __builtin_amdgcn_s_wait_tensorcnt(0);
#else
    if (kb + 32 < 512) {
      _Float16* bnxt = blds + (cur ^ 1) * BUFH;
      for (int i = tid; i < 128 * 4; i += 256) {
        int col = i >> 2, part = i & 3;
        *(v8h*)(bnxt + col * LDB + part * 8) =
            *(const v8h*)(cat + (nbase0 + col) * CATC + kb + 32 + part * 8);
      }
    }
#endif
    __syncthreads();
  }

  int hb = ((lane >> 4) & 1) * 8;
  int nl = lane & 15;
  auto st = [&](v8f acc, int mo, int no) {
#pragma unroll
    for (int r = 0; r < 8; ++r) {
      int o     = mbase + mo * 16 + hb + r;
      long colg = nbase0 + nloc + no * 16 + nl;
      int bb = (int)(colg >> 12), n = (int)(colg & (N_ - 1));
      out[((long)(bb * CATC + o)) * N_ + n] = fmaxf(acc[r], 0.0f);
    }
  };
  st(a00, 0, 0); st(a01, 0, 1); st(a10, 1, 0); st(a11, 1, 1);
}

// ---------------- launcher ----------------
extern "C" void kernel_launch(void* const* d_in, const int* in_sizes, int n_in,
                              void* d_out, int out_size, void* d_ws, size_t ws_size,
                              hipStream_t stream) {
  (void)in_sizes; (void)n_in; (void)out_size; (void)ws_size;
  const float* x  = (const float*)d_in[0];
  const float* w1 = (const float*)d_in[1];
  const float* w2 = (const float*)d_in[2];
  const float* w3 = (const float*)d_in[3];
  const float* w4 = (const float*)d_in[4];
  const float* w5 = (const float*)d_in[5];

  char* ws = (char*)d_ws;
  auto alloc = [&](size_t bytes) {
    char* p = ws;
    ws += (bytes + 255) & ~size_t(255);
    return p;
  };
  _Float16* w1p = (_Float16*)alloc(64 * 32 * 2);
  _Float16* w2h = (_Float16*)alloc(64 * 64 * 2);
  _Float16* w3h = (_Float16*)alloc(128 * 64 * 2);
  _Float16* w4h = (_Float16*)alloc(256 * 128 * 2);
  _Float16* w5h = (_Float16*)alloc(512 * 512 * 2);
  int*      kid = (int*)alloc((size_t)B_ * N_ * KNN * 4);
  _Float16* kds = (_Float16*)alloc((size_t)B_ * N_ * KNN * 2);
  _Float16* catb = (_Float16*)alloc((size_t)B_ * N_ * CATC * 2);

  cvt_weights_kernel<<<(512 * 512 + 255) / 256, 256, 0, stream>>>(
      w1, w2, w3, w4, w5, w1p, w2h, w3h, w4h, w5h);

  size_t knn_smem = 3 * N_ * 4 + 8 * 640 * (4 + 4);          // 90112 B
  knn_kernel<<<B_ * (N_ / 8), 256, knn_smem, stream>>>(x, kid, kds);

  size_t edge_smem = (size_t)2 * NCOL * LDC * 2;              // 87040 B
  edge_kernel<<<B_ * (N_ / PTS), 256, edge_smem, stream>>>(
      x, kid, kds, w1p, w2h, w3h, w4h, catb);

  size_t gemm_smem = (size_t)2 * BUFH * 2;                    // 20480 B
  gemm5_kernel<<<2048, 256, gemm_smem, stream>>>(w5h, catb, (float*)d_out);
}
